// MSDeformableAttention_89678917141181
// MI455X (gfx1250) — matrix operand order repaired and checked by reading, not compiled, not run on hardware
//
#include <hip/hip_runtime.h>
#include <hip/hip_bf16.h>

// ---------------------------------------------------------------------------
// MS-Deformable Attention for MI455X (gfx1250), wave32 + WMMA f16.
// Static problem sizes (match reference):
#define BATCH   4
#define LV      13294          // == Lq
#define CDIM    256
#define NHEADS  8
#define NLEV    4
#define NPTS    4
#define HDIM    32
#define MROWS   (BATCH * LV)   // 53176
#define KDIM    CDIM           // 256 (K of every GEMM)
#define K2DIM   (KDIM / 2)     // 128 packed half2 dwords per row

typedef __attribute__((ext_vector_type(16))) _Float16 v16h;
typedef __attribute__((ext_vector_type(8)))  float    v8f;

union HalfPack2 { _Float16 h[2]; unsigned u; };
union AFrag     { v16h v; _Float16 h[16]; };
union BFrag     { v16h v; unsigned u[8]; };

// ---------------------------------------------------------------------------
// Pack W [K=256, N] (row-major fp32) into B-fragment-friendly f16 layout:
//   Wp[n * 128 + k2] = pack(half(W[2*k2][n]), half(W[2*k2+1][n]))
// B frag (VGPR v(0..3): K = K0 + 8g + 2v{,+1}; v(4..7): +16): lane n%16, group g reads 4
// dwords at k2 = K0/2 + 4g and 4 at +8 -> two coalesced b128 loads/fragment.
__global__ __launch_bounds__(256) void pack_weights(const float* __restrict__ W,
                                                    unsigned* __restrict__ Wp,
                                                    int N) {
    int idx = blockIdx.x * blockDim.x + threadIdx.x;
    if (idx >= N * K2DIM) return;
    int n  = idx >> 7;        // / 128
    int k2 = idx & 127;
    HalfPack2 p;
    p.h[0] = (_Float16)W[(2 * k2 + 0) * N + n];
    p.h[1] = (_Float16)W[(2 * k2 + 1) * N + n];
    Wp[(size_t)n * K2DIM + k2] = p.u;
}

// ---------------------------------------------------------------------------
// GEMM: Out[M, NCOLS] = A[M, 256] (fp32, converted f16 on the fly) * W + bias
// One wave (32 threads) per block; wave computes a 32 x 64 tile:
//   2 M-subtiles x 4 N-subtiles = 8 v_wmma_f32_16x16x32_f16 chains.
// Each B fragment feeds 2 WMMAs, each A fragment feeds 4 -> 2 b128 loads/WMMA.
template <int NCOLS>
__global__ __launch_bounds__(32) void gemm_wmma(const float*    __restrict__ A,
                                                const unsigned* __restrict__ Bp,
                                                const float*    __restrict__ bias,
                                                float*          __restrict__ Out,
                                                int M) {
    constexpr int TILES_N = NCOLS / 64;
    const int mTile = blockIdx.x / TILES_N;
    const int nTile = blockIdx.x % TILES_N;
    const int row0  = mTile * 32;
    const int n0    = nTile * 64;

    const int lane = threadIdx.x;
    const int mIdx = lane & 15;     // M (A) or N (B) index within fragment
    const int g    = lane >> 4;     // lane group

    int rowA = row0 + mIdx;          // rows 0..15 of the tile
    int rowB = row0 + 16 + mIdx;     // rows 16..31 of the tile
    if (rowA >= M) rowA = M - 1;     // clamp: keep EXEC all-1s for WMMA
    if (rowB >= M) rowB = M - 1;

    v8f acc[2][4];
#pragma unroll
    for (int s = 0; s < 2; ++s)
#pragma unroll
        for (int t = 0; t < 4; ++t) acc[s][t] = v8f{};

#pragma unroll
    for (int ks = 0; ks < 8; ++ks) {
        const int K0 = ks * 32;

        // A fragments 16x32 f16: VGPR v(0..3): K = K0+8g+2v{,+1}; v(4..7): +16
        AFrag a0, a1;
        {
            const float* ap0 = A + (size_t)rowA * KDIM + K0 + 8 * g;
            const float* ap1 = ap0 + 16;
            const float* bq0 = A + (size_t)rowB * KDIM + K0 + 8 * g;
            const float* bq1 = bq0 + 16;
#pragma unroll
            for (int i = 0; i < 8; ++i) {
                a0.h[i]     = (_Float16)ap0[i];
                a0.h[8 + i] = (_Float16)ap1[i];
                a1.h[i]     = (_Float16)bq0[i];
                a1.h[8 + i] = (_Float16)bq1[i];
            }
        }

#pragma unroll
        for (int t = 0; t < 4; ++t) {
            // B fragment 32x16 f16: 4 dwords at k2 = K0/2 + 4g, 4 dwords at k2 = K0/2 + 8 + 4g
            const uint4* bp = reinterpret_cast<const uint4*>(
                Bp + (size_t)(n0 + 16 * t + mIdx) * K2DIM + (K0 / 2) + 4 * g);
            BFrag bf;
            uint4 q0 = bp[0];
            uint4 q1 = bp[2];
            bf.u[0] = q0.x; bf.u[1] = q0.y; bf.u[2] = q0.z; bf.u[3] = q0.w;
            bf.u[4] = q1.x; bf.u[5] = q1.y; bf.u[6] = q1.z; bf.u[7] = q1.w;

            acc[0][t] = __builtin_amdgcn_wmma_f32_16x16x32_f16(
                false, a0.v, false, bf.v, (short)0, acc[0][t], false, false);
            acc[1][t] = __builtin_amdgcn_wmma_f32_16x16x32_f16(
                false, a1.v, false, bf.v, (short)0, acc[1][t], false, false);
        }
    }

    // C/D layout: VGPR j, lanes 0-15 -> M=j, lanes 16-31 -> M=j+8; N = lane&15
#pragma unroll
    for (int s = 0; s < 2; ++s) {
#pragma unroll
        for (int t = 0; t < 4; ++t) {
            const int n = n0 + 16 * t + mIdx;
            const float bn = bias[n];
#pragma unroll
            for (int j = 0; j < 8; ++j) {
                const int m = row0 + 16 * s + j + 8 * g;
                if (m < M) Out[(size_t)m * NCOLS + n] = acc[s][t][j] + bn;
            }
        }
    }
}

// ---------------------------------------------------------------------------
// Fused softmax + bilinear sampling + attention-weighted accumulation.
// One wave per (query, head); lane = channel d (HDIM == 32 == wave width),
// so every corner gather is one coalesced 128B global load (L2-resident).
__global__ __launch_bounds__(256) void msda_sample(const float* __restrict__ vproj,
                                                   const float* __restrict__ offr,
                                                   const float* __restrict__ attr,
                                                   const float* __restrict__ refp,
                                                   float*       __restrict__ tmp) {
    const int m    = blockIdx.x;          // b*Lq + q
    const int head = threadIdx.x >> 5;
    const int lane = threadIdx.x & 31;
    const int b    = m / LV;

    const int Hs[NLEV]     = {100, 50, 25, 13};
    const int Ws[NLEV]     = {100, 50, 25, 13};
    const int starts[NLEV] = {0, 10000, 12500, 13125};

    // softmax over L*P = 16 (uniform across the wave)
    const float* ar = attr + (size_t)m * (NHEADS * NLEV * NPTS) + head * (NLEV * NPTS);
    float w[16];
    float mx = -1e30f;
#pragma unroll
    for (int i = 0; i < 16; ++i) { w[i] = ar[i]; mx = fmaxf(mx, w[i]); }
    float s = 0.f;
#pragma unroll
    for (int i = 0; i < 16; ++i) { w[i] = __expf(w[i] - mx); s += w[i]; }
    const float inv = 1.f / s;

    const float* orow  = offr + (size_t)m * CDIM + head * (NLEV * NPTS * 2);
    const float* vbase = vproj + ((size_t)b * LV) * CDIM + head * HDIM + lane;

    float acc = 0.f;
#pragma unroll
    for (int l = 0; l < NLEV; ++l) {
        const int   Hl = Hs[l], Wl = Ws[l], st = starts[l];
        const float rx = refp[((size_t)m * NLEV + l) * 2 + 0];
        const float ry = refp[((size_t)m * NLEV + l) * 2 + 1];
#pragma unroll
        for (int p = 0; p < NPTS; ++p) {
            const float ox = orow[(l * NPTS + p) * 2 + 0];
            const float oy = orow[(l * NPTS + p) * 2 + 1];
            // loc = ref + off/(W,H); x = loc_x*W - 0.5 = ref_x*W + off_x - 0.5
            const float x = rx * (float)Wl + ox - 0.5f;
            const float y = ry * (float)Hl + oy - 0.5f;
            const float x0f = floorf(x), y0f = floorf(y);
            const int   x0 = (int)x0f, y0 = (int)y0f;
            const float fx = x - x0f, fy = y - y0f;

            float v00 = 0.f, v10 = 0.f, v01 = 0.f, v11 = 0.f;
            const bool xv0 = (x0 >= 0) && (x0 < Wl);
            const bool xv1 = (x0 + 1 >= 0) && (x0 + 1 < Wl);
            const bool yv0 = (y0 >= 0) && (y0 < Hl);
            const bool yv1 = (y0 + 1 >= 0) && (y0 + 1 < Hl);
            if (yv0) {
                const size_t rowb = (size_t)(st + y0 * Wl);
                if (xv0) v00 = vbase[(rowb + x0) * CDIM];
                if (xv1) v10 = vbase[(rowb + x0 + 1) * CDIM];
            }
            if (yv1) {
                const size_t rowb = (size_t)(st + (y0 + 1) * Wl);
                if (xv0) v01 = vbase[(rowb + x0) * CDIM];
                if (xv1) v11 = vbase[(rowb + x0 + 1) * CDIM];
            }
            const float bil = (1.f - fx) * (1.f - fy) * v00 + fx * (1.f - fy) * v10 +
                              (1.f - fx) * fy * v01 + fx * fy * v11;
            acc += (w[l * NPTS + p] * inv) * bil;
        }
    }
    tmp[(size_t)m * CDIM + head * HDIM + lane] = acc;
}

// ---------------------------------------------------------------------------
extern "C" void kernel_launch(void* const* d_in, const int* in_sizes, int n_in,
                              void* d_out, int out_size, void* d_ws, size_t ws_size,
                              hipStream_t stream) {
    const float* query = (const float*)d_in[0];
    const float* refp  = (const float*)d_in[1];
    const float* value = (const float*)d_in[2];
    // d_in[3]: value_spatial_shapes — static in reference, hardcoded above.
    const float* W_v   = (const float*)d_in[4];
    const float* b_v   = (const float*)d_in[5];
    const float* W_off = (const float*)d_in[6];
    const float* b_off = (const float*)d_in[7];
    const float* W_att = (const float*)d_in[8];
    const float* b_att = (const float*)d_in[9];
    const float* W_out = (const float*)d_in[10];
    const float* b_out = (const float*)d_in[11];
    float* out = (float*)d_out;

    // Workspace carve-up (all sizes multiples of 256B)
    char* ws = (char*)d_ws;
    unsigned* Wv_p   = (unsigned*)ws; ws += (size_t)CDIM * K2DIM * 4;          // 128 KB
    unsigned* Woff_p = (unsigned*)ws; ws += (size_t)CDIM * K2DIM * 4;          // 128 KB
    unsigned* Watt_p = (unsigned*)ws; ws += (size_t)128  * K2DIM * 4;          //  64 KB
    unsigned* Wout_p = (unsigned*)ws; ws += (size_t)CDIM * K2DIM * 4;          // 128 KB
    float* vproj = (float*)ws; ws += (size_t)MROWS * CDIM * 4;                 // 54.5 MB
    float* offr  = (float*)ws; ws += (size_t)MROWS * CDIM * 4;                 // 54.5 MB
    float* attr  = (float*)ws; ws += (size_t)MROWS * 128  * 4;                 // 27.2 MB
    float* tmp   = (float*)ws;                                                 // 54.5 MB

    // 1) pack weights into f16 B-fragment layout
    pack_weights<<<(CDIM * K2DIM + 255) / 256, 256, 0, stream>>>(W_v,   Wv_p,   CDIM);
    pack_weights<<<(CDIM * K2DIM + 255) / 256, 256, 0, stream>>>(W_off, Woff_p, CDIM);
    pack_weights<<<(128  * K2DIM + 255) / 256, 256, 0, stream>>>(W_att, Watt_p, 128);
    pack_weights<<<(CDIM * K2DIM + 255) / 256, 256, 0, stream>>>(W_out, Wout_p, CDIM);

    const int mTiles32 = (MROWS + 31) / 32;   // 1662

    // 2) value projection, sampling offsets, attention logits (WMMA GEMMs)
    gemm_wmma<256><<<dim3(mTiles32 * 4), 32, 0, stream>>>(value, Wv_p,   b_v,   vproj, MROWS);
    gemm_wmma<256><<<dim3(mTiles32 * 4), 32, 0, stream>>>(query, Woff_p, b_off, offr,  MROWS);
    gemm_wmma<128><<<dim3(mTiles32 * 2), 32, 0, stream>>>(query, Watt_p, b_att, attr,  MROWS);

    // 3) fused softmax + bilinear gather + weighted sum
    msda_sample<<<dim3(MROWS), 256, 0, stream>>>(vproj, offr, attr, refp, tmp);

    // 4) output projection
    gemm_wmma<256><<<dim3(mTiles32 * 4), 32, 0, stream>>>(tmp, Wout_p, b_out, out, MROWS);
}
